// TriModalCrossAttention_ver2_11484742549629
// MI455X (gfx1250) — matrix operand order repaired and checked by reading, not compiled, not run on hardware
//
#include <hip/hip_runtime.h>
#include <hip/hip_bf16.h>

typedef __bf16 bf16_t;
typedef __attribute__((ext_vector_type(16))) __bf16 v16bf;
typedef __attribute__((ext_vector_type(8)))  float  v8f;

#define BM 128
#define BN 128
#define BK 32
#define LDSTR 40   // bf16 elems per LDS row: 32 + 8 pad; row = 80B (16B aligned)

union FragBF { v16bf v; unsigned u[8]; };

// ---------------------------------------------------------------------------
// CDNA5 async direct global->LDS copy (no VGPR bounce), tracked by ASYNCcnt.
// lds_off: per-lane LDS byte offset (addrspace(3) value = low 32 bits of the
// flat pointer for __shared__ objects). g: 16B-aligned global address.
// ---------------------------------------------------------------------------
__device__ inline unsigned lds_off_of(const void* p) {
  return (unsigned)(unsigned long long)p;   // LDS aperture: addr[31:0] = LDS offset
}
__device__ inline void async_copy_b128(unsigned lds_off, const void* g) {
  asm volatile("global_load_async_to_lds_b128 %0, %1, off"
               :: "v"(lds_off), "v"(g) : "memory");
}
__device__ inline void wait_async0() {
  asm volatile("s_wait_asynccnt 0x0" ::: "memory");
}

// ---------------------------------------------------------------------------
// fp32 -> bf16 conversion with strided destination (builds the bf16 concat
// buffer Xc[:,3D] and bf16 weight copies).
// ---------------------------------------------------------------------------
__global__ __launch_bounds__(256) void cvt_f32_bf16_kernel(
    const float* __restrict__ src, bf16_t* __restrict__ dst,
    long long n, int src_cols, int dst_ld, int dst_off)
{
  long long i = (long long)blockIdx.x * blockDim.x + threadIdx.x;
  if (i >= n) return;
  long long r = i / src_cols;
  int c = (int)(i - r * src_cols);
  dst[r * (long long)dst_ld + dst_off + c] = (bf16_t)src[i];
}

// ---------------------------------------------------------------------------
// Row softmax: one 256-thread block per row; reads f32 scores, writes bf16 P.
// ---------------------------------------------------------------------------
__global__ __launch_bounds__(256) void softmax_rows_kernel(
    const float* __restrict__ scores, bf16_t* __restrict__ P, int ncols)
{
  long long row = blockIdx.x;
  const float* src = scores + row * (long long)ncols;
  bf16_t* dst = P + row * (long long)ncols;
  int tid = threadIdx.x;
  __shared__ float red[256];

  float lmax = -1e30f;
  for (int c = tid; c < ncols; c += 256) lmax = fmaxf(lmax, src[c]);
  red[tid] = lmax; __syncthreads();
  for (int s = 128; s > 0; s >>= 1) {
    if (tid < s) red[tid] = fmaxf(red[tid], red[tid + s]);
    __syncthreads();
  }
  float m = red[0]; __syncthreads();

  float lsum = 0.f;
  for (int c = tid; c < ncols; c += 256) lsum += __expf(src[c] - m);
  red[tid] = lsum; __syncthreads();
  for (int s = 128; s > 0; s >>= 1) {
    if (tid < s) red[tid] += red[tid + s];
    __syncthreads();
  }
  float inv = 1.0f / red[0];
  for (int c = tid; c < ncols; c += 256) dst[c] = (bf16_t)(__expf(src[c] - m) * inv);
}

// ---------------------------------------------------------------------------
// Generic bf16 WMMA GEMM:   C = alpha * A @ op(B) + bias (+ C if accum)
//   A: [M,K] bf16 row-major (lda)
//   B: NN -> [K,N] row-major (ldb);  NT (bnt=1) -> [N,K] row-major (A@B^T)
//   Cf : optional f32 dest (ldc)       Cb : optional bf16 dest (ldcb)
//   Cf2: optional second f32 dest, interleaved row mapping (global_feature):
//        grow = (row/Sseq)*3*Sseq + (row%Sseq)*gstride + goff
// Macro tile 128x128x32, 256 threads = 8 wave32 waves, each wave 4x2 WMMA
// tiles. Staging uses GLOBAL_LOAD_ASYNC_TO_LDS_B128 where no transpose is
// needed. M,N multiples of 128; K multiple of 32; lda/ldb multiples of 8.
// ---------------------------------------------------------------------------
__global__ __launch_bounds__(256) void gemm_bf16_kernel(
    const bf16_t* __restrict__ A, int lda,
    const bf16_t* __restrict__ Bm, int ldb, int bnt,
    const float* __restrict__ bias,
    float* __restrict__ Cf, int ldc,
    bf16_t* __restrict__ Cb, int ldcb,
    float* __restrict__ Cf2, int Sseq, int gstride, int goff,
    int M, int N, int K, float alpha, int accum)
{
  __shared__ bf16_t As[BM * LDSTR];
  __shared__ bf16_t Bs[BN * LDSTR];

  const int tid  = threadIdx.x;
  const int lane = tid & 31;
  const int wave = tid >> 5;
  const int l = lane & 15;        // lane%16
  const int h = lane >> 4;        // lane half
  const int wm = wave >> 2;       // 0..1 : 64-row slab
  const int wn = wave & 3;        // 0..3 : 32-col slab
  const long long m0 = (long long)blockIdx.y * BM;
  const long long n0 = (long long)blockIdx.x * BN;

  v8f acc[4][2];
#pragma unroll
  for (int mt = 0; mt < 4; ++mt)
#pragma unroll
    for (int nt = 0; nt < 2; ++nt) { v8f z = {}; acc[mt][nt] = z; }

  const int ar  = tid >> 2;          // 0..63 row within tile
  const int ac  = (tid & 3) * 8;     // 0,8,16,24 k-offset (8 bf16 = 16B)
  const int bkk = tid >> 4;          // 0..15 k-row for NN transpose copy
  const int bkn = (tid & 15) * 8;    // 0..120 n-offset

  // per-thread LDS destinations for the async copies (constant across k0)
  const unsigned lAs0 = lds_off_of(&As[(ar      ) * LDSTR + ac]);
  const unsigned lAs1 = lds_off_of(&As[(ar + 64 ) * LDSTR + ac]);
  const unsigned lBs0 = lds_off_of(&Bs[(ar      ) * LDSTR + ac]);
  const unsigned lBs1 = lds_off_of(&Bs[(ar + 64 ) * LDSTR + ac]);

  for (int k0 = 0; k0 < K; k0 += BK) {
    // ---- stage A tile 128x32 into LDS via async direct-to-LDS copies ----
    async_copy_b128(lAs0, A + (m0 + ar      ) * (long long)lda + k0 + ac);
    async_copy_b128(lAs1, A + (m0 + ar + 64 ) * (long long)lda + k0 + ac);
    // ---- stage B tile into LDS as [n][k] ----
    if (bnt) {           // B is [N,K]: straight async copy
      async_copy_b128(lBs0, Bm + (n0 + ar      ) * (long long)ldb + k0 + ac);
      async_copy_b128(lBs1, Bm + (n0 + ar + 64 ) * (long long)ldb + k0 + ac);
    } else {             // B is [K,N]: transpose during the copy (VGPR bounce)
#pragma unroll
      for (int it = 0; it < 2; ++it) {
        int kr = bkk + it * 16;
        union { uint4 q; bf16_t e[8]; } t;
        t.q = *(const uint4*)(Bm + (long long)(k0 + kr) * ldb + n0 + bkn);
#pragma unroll
        for (int j = 0; j < 8; ++j) Bs[(bkn + j) * LDSTR + kr] = t.e[j];
      }
      if (k0 + BK < K)
        __builtin_prefetch(Bm + (long long)(k0 + BK + bkk) * ldb + n0 + bkn, 0, 3);
    }
    wait_async0();       // ASYNCcnt == 0 : async tiles resident in LDS
    __syncthreads();

    // ---- load fragments per ISA 7.12.2 layouts ----
    FragBF fa[4], fb[2];
#pragma unroll
    for (int mt = 0; mt < 4; ++mt) {
      const unsigned* p = (const unsigned*)&As[(wm * 64 + mt * 16 + l) * LDSTR];
#pragma unroll
      for (int j = 0; j < 8; ++j) {
        // A: vgpr j<4 holds K=2j+8h pair; j>=4 holds K=16+2(j-4)+8h pair
        int idx = (j < 4) ? (j + 4 * h) : (j + 4 + 4 * h);
        fa[mt].u[j] = p[idx];
      }
    }
#pragma unroll
    for (int nt = 0; nt < 2; ++nt) {
      const unsigned* p = (const unsigned*)&Bs[(wn * 32 + nt * 16 + l) * LDSTR];
#pragma unroll
      for (int j = 0; j < 8; ++j) fb[nt].u[j] = p[(j < 4) ? (j + 4 * h) : (j + 4 + 4 * h)];  // B: same K pairs as A
    }

#pragma unroll
    for (int mt = 0; mt < 4; ++mt)
#pragma unroll
      for (int nt = 0; nt < 2; ++nt)
        acc[mt][nt] = __builtin_amdgcn_wmma_f32_16x16x32_bf16(
            false, fa[mt].v, false, fb[nt].v, (short)0, acc[mt][nt], false, false);
    __syncthreads();
  }

  // ---- epilogue: alpha, bias, accumulate, f32/bf16/interleaved stores ----
#pragma unroll
  for (int mt = 0; mt < 4; ++mt) {
#pragma unroll
    for (int nt = 0; nt < 2; ++nt) {
#pragma unroll
      for (int g = 0; g < 8; ++g) {
        long long row = m0 + wm * 64 + mt * 16 + g + 8 * h;   // C/D: M = g + 8h
        long long col = n0 + wn * 32 + nt * 16 + l;           //       N = lane%16
        float v = acc[mt][nt][g] * alpha;
        if (bias)  v += bias[col];
        if (accum) v += Cf[row * ldc + col];
        if (Cf)  Cf[row * ldc + col] = v;
        if (Cb)  Cb[row * ldcb + col] = (bf16_t)v;
        if (Cf2) {
          long long b = row / Sseq;
          long long s = row - b * Sseq;
          long long grow = b * 3LL * Sseq + s * gstride + goff;
          Cf2[grow * ldc + col] = v;
        }
      }
    }
  }
}

// ---------------------------------------------------------------------------
// Host orchestration
// ---------------------------------------------------------------------------
extern "C" void kernel_launch(void* const* d_in, const int* in_sizes, int n_in,
                              void* d_out, int out_size, void* d_ws, size_t ws_size,
                              hipStream_t stream)
{
  (void)in_sizes; (void)n_in; (void)out_size; (void)ws_size;
  const int D = 1024, Bt = 4, S = 2048;
  const int M = Bt * S;  // 8192 rows (batches concatenated)

  const float* x[3]  = {(const float*)d_in[0], (const float*)d_in[1], (const float*)d_in[2]};
  const float* Wq    = (const float*)d_in[3];
  const float* bq    = (const float*)d_in[4];
  const float* Wk[3] = {(const float*)d_in[5],  (const float*)d_in[9],  (const float*)d_in[13]};
  const float* bk[3] = {(const float*)d_in[6],  (const float*)d_in[10], (const float*)d_in[14]};
  const float* Wv[3] = {(const float*)d_in[7],  (const float*)d_in[11], (const float*)d_in[15]};
  const float* bv[3] = {(const float*)d_in[8],  (const float*)d_in[12], (const float*)d_in[16]};
  const float* Wo[3] = {(const float*)d_in[17], (const float*)d_in[19], (const float*)d_in[21]};
  const float* bo[3] = {(const float*)d_in[18], (const float*)d_in[20], (const float*)d_in[22]};

  float* out = (float*)d_out;
  float* gf  = out + 3LL * M * D;   // global_feature [B, 3S, D]

  // workspace carve-out (~265 MB total)
  char* ws = (char*)d_ws; size_t off = 0;
  auto alloc = [&](size_t bytes) -> char* {
    char* p = ws + off; off += (bytes + 255) & ~(size_t)255; return p;
  };
  bf16_t* Xc   = (bf16_t*)alloc((size_t)M * 3 * D * 2);   // bf16 concat [M, 3D]
  bf16_t* Wq_b = (bf16_t*)alloc((size_t)3 * D * D * 2);
  bf16_t *Wk_b[3], *Wv_b[3], *Wo_b[3], *Kb[3], *Vb[3], *Ctx[3];
  for (int i = 0; i < 3; ++i) {
    Wk_b[i] = (bf16_t*)alloc((size_t)D * D * 2);
    Wv_b[i] = (bf16_t*)alloc((size_t)D * D * 2);
    Wo_b[i] = (bf16_t*)alloc((size_t)2 * D * D * 2);
  }
  bf16_t* Qb = (bf16_t*)alloc((size_t)M * D * 2);
  for (int i = 0; i < 3; ++i) {
    Kb[i]  = (bf16_t*)alloc((size_t)M * D * 2);
    Vb[i]  = (bf16_t*)alloc((size_t)M * D * 2);
    Ctx[i] = (bf16_t*)alloc((size_t)M * D * 2);
  }
  float*  Sc = (float*)alloc((size_t)S * S * 4);   // one scores matrix (reused x12)
  bf16_t* P  = (bf16_t*)alloc((size_t)S * S * 2);  // one prob matrix (reused x12)

  auto cvt = [&](const float* s, bf16_t* d, long long n, int cols, int ld, int o) {
    cvt_f32_bf16_kernel<<<(unsigned)((n + 255) / 256), 256, 0, stream>>>(s, d, n, cols, ld, o);
  };
  auto gemm = [&](const bf16_t* A, int lda, const bf16_t* Bm, int ldb, int bnt,
                  const float* bias, float* Cf, int ldc, bf16_t* Cb, int ldcb,
                  float* Cf2, int gstride, int goff,
                  int Mm, int Nn, int Kk, float alpha, int accum) {
    dim3 grid(Nn / BN, Mm / BM);
    gemm_bf16_kernel<<<grid, 256, 0, stream>>>(A, lda, Bm, ldb, bnt, bias, Cf, ldc,
                                               Cb, ldcb, Cf2, S, gstride, goff,
                                               Mm, Nn, Kk, alpha, accum);
  };

  // 1) bf16 conversions: x -> Xc columns, weights -> bf16 copies
  for (int i = 0; i < 3; ++i) cvt(x[i], Xc, (long long)M * D, D, 3 * D, i * D);
  cvt(Wq, Wq_b, 3LL * D * D, D, D, 0);
  for (int i = 0; i < 3; ++i) {
    cvt(Wk[i], Wk_b[i], (long long)D * D, D, D, 0);
    cvt(Wv[i], Wv_b[i], (long long)D * D, D, D, 0);
    cvt(Wo[i], Wo_b[i], 2LL * D * D, D, D, 0);
  }

  // 2) Q = concat(x1,x2,x3) @ Wq + bq   (bf16 out, feeds QK^T)
  gemm(Xc, 3 * D, Wq_b, D, 0, bq, nullptr, 0, Qb, D, nullptr, 0, 0, M, D, 3 * D, 1.0f, 0);

  // 3) K_i = x_i@Wk_i + bk ; V_i = x_i@Wv_i + bv   (A = strided column slice of Xc)
  for (int i = 0; i < 3; ++i) {
    gemm(Xc + i * D, 3 * D, Wk_b[i], D, 0, bk[i], nullptr, 0, Kb[i], D, nullptr, 0, 0, M, D, D, 1.0f, 0);
    gemm(Xc + i * D, 3 * D, Wv_b[i], D, 0, bv[i], nullptr, 0, Vb[i], D, nullptr, 0, 0, M, D, D, 1.0f, 0);
  }

  // 4) attention per (batch, modality): scores = Q K^T / sqrt(D); softmax; ctx = P V
  const float scale = 1.0f / 32.0f;  // 1/sqrt(1024)
  for (int b = 0; b < Bt; ++b) {
    for (int i = 0; i < 3; ++i) {
      gemm(Qb + (long long)b * S * D, D, Kb[i] + (long long)b * S * D, D, /*NT=*/1,
           nullptr, Sc, S, nullptr, 0, nullptr, 0, 0, S, S, D, scale, 0);
      softmax_rows_kernel<<<S, 256, 0, stream>>>(Sc, P, S);
      gemm(P, S, Vb[i] + (long long)b * S * D, D, 0,
           nullptr, nullptr, 0, Ctx[i] + (long long)b * S * D, D, nullptr, 0, 0,
           S, D, S, 1.0f, 0);
    }
  }

  // 5) out_i = x_i@Wo_i[:D] + ctx_i@Wo_i[D:] + bo_i  (two accumulating GEMMs;
  //    second pass also scatters rows into global_feature: interleave/append)
  for (int i = 0; i < 3; ++i) {
    float* Ci = out + (long long)i * M * D;
    gemm(Xc + i * D, 3 * D, Wo_b[i], D, 0, bo[i], Ci, D, nullptr, 0, nullptr, 0, 0,
         M, D, D, 1.0f, 0);
    int gs = (i < 2) ? 2 : 1;
    int go = (i == 0) ? 0 : (i == 1) ? 1 : 2 * S;
    gemm(Ctx[i], D, Wo_b[i] + (size_t)D * D, D, 0, nullptr, Ci, D, nullptr, 0,
         gf, gs, go, M, D, D, 1.0f, 1);
  }
}
